// AttnAdaIN_163208757939
// MI455X (gfx1250) — matrix operand order repaired and checked by reading, not compiled, not run on hardware
//
#include <hip/hip_runtime.h>
#include <cmath>

// ---------------- problem constants (from reference shapes) ----------------
constexpr int Bn = 4;     // batch
constexpr int C  = 512;   // channels (= key channels)
constexpr int N  = 4096;  // H*W content
constexpr int M  = 4096;  // H*W style

// ---------------- WMMA types ----------------
typedef __attribute__((ext_vector_type(16))) __bf16 v16bf;
typedef __attribute__((ext_vector_type(8)))  float  v8f;

// pointee type expected by the async-to-LDS builtins (from hipcc diagnostic)
typedef __attribute__((__vector_size__(4 * sizeof(int)))) int vsi4;
typedef __attribute__((address_space(1))) vsi4* gv4i_ptr;   // global v4i*
typedef __attribute__((address_space(3))) vsi4* lv4i_ptr;   // LDS v4i*

// ---------------- feature probes (compile-safe fallbacks) ----------------
#if defined(__has_builtin)
#  if __has_builtin(__builtin_amdgcn_global_load_async_to_lds_b128)
#    define HAVE_ASYNC_COPY 1
#  else
#    define HAVE_ASYNC_COPY 0
#  endif
#  if __has_builtin(__builtin_amdgcn_cvt_pk_bf16_f32)
#    define HAVE_PK_BF16 1
#  else
#    define HAVE_PK_BF16 0
#  endif
#else
#  define HAVE_ASYNC_COPY 0
#  define HAVE_PK_BF16 0
#endif

__device__ __forceinline__ unsigned short f2bf(float x) {
    unsigned u = __float_as_uint(x);
    unsigned r = u + 0x7fffu + ((u >> 16) & 1u);   // round-to-nearest-even
    return (unsigned short)(r >> 16);
}
__device__ __forceinline__ float bf2f(unsigned short h) {
    return __uint_as_float(((unsigned)h) << 16);
}
// packed f32x2 -> bf16x2 (1 VALU op on CDNA5 if the builtin exists)
__device__ __forceinline__ unsigned pk_bf16(float lo, float hi) {
#if HAVE_PK_BF16
    typedef __attribute__((ext_vector_type(2))) __bf16 v2bf;
    union { v2bf v; unsigned u; } cv;
    cv.v = __builtin_amdgcn_cvt_pk_bf16_f32(lo, hi);
    return cv.u;
#else
    return (unsigned)f2bf(lo) | ((unsigned)f2bf(hi) << 16);
#endif
}

// 16-byte global -> LDS copy: gfx1250 async path (ASYNCcnt), sync fallback.
__device__ __forceinline__ void copy16(const unsigned short* g, unsigned short* l) {
#if HAVE_ASYNC_COPY
    __builtin_amdgcn_global_load_async_to_lds_b128((gv4i_ptr)g, (lv4i_ptr)l, 0, 0);
#else
    *(uint4*)l = *(const uint4*)g;
#endif
}
__device__ __forceinline__ void async_wait0() {
#if HAVE_ASYNC_COPY
#  if __has_builtin(__builtin_amdgcn_s_wait_asynccnt)
    __builtin_amdgcn_s_wait_asynccnt(0);
#  else
    asm volatile("s_wait_asynccnt 0" ::: "memory");
#  endif
#endif
}

__device__ __forceinline__ v8f wmma_bf16(v16bf a, v16bf b, v8f c) {
    return __builtin_amdgcn_wmma_f32_16x16x32_bf16(false, a, false, b, (short)0, c, false, false);
}

// A-matrix 16x32 bf16 fragment (ISA 7.12.2): lane half 0 -> K 0..7 (v0..3), 16..23 (v4..7);
// lane half 1 -> K 8..15, 24..31.  Pairs contiguous along the LDS row.
__device__ __forceinline__ v16bf load_frag_a(const unsigned short* lds, int row, int rowStride,
                                             int kbase, int lh) {
    union { v16bf v; unsigned u[8]; } f;
    const unsigned* p = (const unsigned*)(lds + row * rowStride + kbase);
#pragma unroll
    for (int i = 0; i < 8; ++i) {
        int k = (i < 4 ? 2 * i : 16 + 2 * (i - 4)) + 8 * lh;
        f.u[i] = p[k >> 1];
    }
    return f.v;
}

// B-matrix 32x16 bf16 fragment: lanes 0-15 hold K=0..7,16..23; lanes 16-31 hold K=8..15,24..31.
__device__ __forceinline__ v16bf load_frag_b(const unsigned short* lds, int row, int rowStride,
                                             int kbase, int lh) {
    union { v16bf v; unsigned u[8]; } f;
    const unsigned* p = (const unsigned*)(lds + row * rowStride + kbase);
#pragma unroll
    for (int i = 0; i < 8; ++i) {
        int k = (i < 4 ? 2 * i : 16 + 2 * (i - 4)) + 8 * lh;
        f.u[i] = p[k >> 1];
    }
    return f.v;
}

// ============================================================================
// Kernel 1: 1x1-conv GEMM.  out[b, n, co] = sum_c W[co,c] * x[b,c,n] + bias[co]
// A = x^T (N x C), B = W^T (C x Cout).  Output bf16 in [b][n][co] row-major.
// Block 128 (4 waves), tile 64n x 64co, K staged 32 at a time in LDS.
// ============================================================================
__global__ __launch_bounds__(128) void proj_kernel(const float* __restrict__ X,
                                                   const float* __restrict__ Wt,
                                                   const float* __restrict__ bias,
                                                   unsigned short* __restrict__ Out) {
    __shared__ unsigned short xs[64 * 32];  // [n][c]
    __shared__ unsigned short wsh[64 * 32]; // [co][c]

    const int tid  = threadIdx.x;
    const int wave = tid >> 5;
    const int lane = tid & 31;
    const int lh   = lane >> 4;
    const int l16  = lane & 15;

    const int n0  = blockIdx.x * 64;
    const int co0 = blockIdx.y * 64;
    const int b   = blockIdx.z;

    const float* Xb = X + (size_t)b * C * N;

    v8f acc[4];
#pragma unroll
    for (int t = 0; t < 4; ++t) acc[t] = (v8f)0.0f;

    for (int kk = 0; kk < C / 32; ++kk) {
        const int c0 = kk * 32;
        // stage x tile [32c x 64n] -> xs[n][c] bf16 (transpose during store)
#pragma unroll
        for (int i = 0; i < 4; ++i) {
            int f   = tid + i * 128;     // 0..511 float4s
            int row = f >> 4;            // c row (0..31)
            int cg  = f & 15;            // n group of 4
            float4 x4 = *(const float4*)(Xb + (size_t)(c0 + row) * N + n0 + cg * 4);
            int nb = cg * 4;
            xs[(nb + 0) * 32 + row] = f2bf(x4.x);
            xs[(nb + 1) * 32 + row] = f2bf(x4.y);
            xs[(nb + 2) * 32 + row] = f2bf(x4.z);
            xs[(nb + 3) * 32 + row] = f2bf(x4.w);
        }
        // stage W tile [64co x 32c] -> wsh[co][c] bf16 (packed pair stores)
#pragma unroll
        for (int i = 0; i < 4; ++i) {
            int f   = tid + i * 128;     // 0..511 float4s
            int row = f >> 3;            // co row (0..63)
            int cg  = f & 7;             // c group of 4
            float4 w4 = *(const float4*)(Wt + (size_t)(co0 + row) * C + c0 + cg * 4);
            unsigned* dst = (unsigned*)&wsh[row * 32 + cg * 4];
            dst[0] = pk_bf16(w4.x, w4.y);
            dst[1] = pk_bf16(w4.z, w4.w);
        }
        __syncthreads();

        v16bf afr = load_frag_a(xs, wave * 16 + l16, 32, 0, lh);
#pragma unroll
        for (int t = 0; t < 4; ++t) {
            v16bf bfr = load_frag_b(wsh, t * 16 + l16, 32, 0, lh);
            acc[t] = wmma_bf16(afr, bfr, acc[t]);
        }
        __syncthreads();
    }

    // epilogue: +bias, convert to bf16, store [b][n][co]
    unsigned short* Ob = Out + (size_t)b * N * C;
#pragma unroll
    for (int t = 0; t < 4; ++t) {
        int co = co0 + t * 16 + l16;
        float bv = bias[co];
#pragma unroll
        for (int e = 0; e < 8; ++e) {
            int n = n0 + wave * 16 + e + 8 * lh;
            Ob[(size_t)n * C + co] = f2bf(acc[t][e] + bv);
        }
    }
}

// ============================================================================
// Kernel 2: fused flash attention with two accumulators (mean & second moment).
// Workgroup: 64 query rows, 16 waves (512 thr).  M-tile = 32.
//  - Q/K tiles staged with gfx1250 async global->LDS (ASYNCcnt) when available
//  - V staged transposed; V^2 pre-squared once into a second LDS tile
//  - waves 0-7: one 16x16 logit tile each over full Ck=512
//  - all 16 waves: 16 persistent f32 accumulator tiles (P*V and P*V^2)
// LDS ~173 KB (CDNA5 WGP has 320 KB).
// ============================================================================
__global__ __launch_bounds__(512) void attn_kernel(const unsigned short* __restrict__ Fq,
                                                   const unsigned short* __restrict__ Gk,
                                                   const unsigned short* __restrict__ Vv,
                                                   float* __restrict__ outMean,
                                                   float* __restrict__ outStd) {
    __shared__ unsigned short Qs[64 * 512];  // [n][ck]
    __shared__ unsigned short Ks[32 * 512];  // [m][ck]
    __shared__ unsigned short Vt[512 * 32];  // [c][m]   (transposed for B-frag pairs)
    __shared__ unsigned short Vt2[512 * 32]; // [c][m]   (elementwise squared)
    __shared__ float          Sf[64 * 32];   // logits tile
    __shared__ unsigned short Ps[64 * 32];   // exp(P) bf16
    __shared__ float rowM[64], rowL[64], rowScale[64];

    const int tid  = threadIdx.x;
    const int wave = tid >> 5;
    const int lane = tid & 31;
    const int lh   = lane >> 4;
    const int l16  = lane & 15;

    const int n0 = blockIdx.x * 64;
    const int b  = blockIdx.y;

    const int rb    = wave & 3;          // row block (16 rows of 64)
    const int cbase = (wave >> 2) * 128; // 128-wide channel slice -> 8 col tiles

    v8f accM[8], accS[8];
#pragma unroll
    for (int t = 0; t < 8; ++t) { accM[t] = (v8f)0.0f; accS[t] = (v8f)0.0f; }

    if (tid < 64) { rowM[tid] = -INFINITY; rowL[tid] = 0.0f; }

    // stage Q once (contiguous, async if available): 64*512 bf16
    {
        const unsigned short* src = Fq + ((size_t)b * N + n0) * C;
        for (int i = tid; i < 64 * 512 / 8; i += 512) copy16(src + i * 8, Qs + i * 8);
    }

    for (int j0 = 0; j0 < M; j0 += 32) {
        __syncthreads();  // previous iteration's consumers are done with Ks/Vt/Ps
        // ---- stage K tile (async contiguous copy) ----
        {
            const unsigned short* src = Gk + ((size_t)b * M + j0) * C;
            for (int i = tid; i < 32 * 512 / 8; i += 512) copy16(src + i * 8, Ks + i * 8);
        }
        // ---- stage V tile transposed + pre-squared ----
        for (int i = tid; i < 32 * 512 / 8; i += 512) {
            int m  = i >> 6;             // 64 uint4 per m-row
            int c8 = (i & 63) * 8;
            union { uint4 q; unsigned short s[8]; } u;
            u.q = *(const uint4*)(Vv + ((size_t)b * M + j0 + m) * C + c8);
#pragma unroll
            for (int e = 0; e < 8; ++e) {
                unsigned short h = u.s[e];
                Vt[(c8 + e) * 32 + m] = h;
                float fv = bf2f(h);
                Vt2[(c8 + e) * 32 + m] = f2bf(fv * fv);
            }
        }
        // ---- prefetch next tiles into L2 while we compute this one ----
        if (j0 + 32 < M) {
            const unsigned short* nk = Gk + ((size_t)b * M + j0 + 32) * C + (size_t)tid * 32;
            const unsigned short* nv = Vv + ((size_t)b * M + j0 + 32) * C + (size_t)tid * 32;
            __builtin_prefetch(nk, 0, 1);
            __builtin_prefetch(nv, 0, 1);
        }
        async_wait0();
        __syncthreads();

        // ---- logits: S(64x32) = Q(64x512) * K^T, waves 0..7 (one tile each) ----
        if (wave < 8) {
            const int srb = wave & 3, scb = wave >> 2;
            v8f s = (v8f)0.0f;
            for (int kk = 0; kk < C; kk += 32) {
                v16bf afr = load_frag_a(Qs, srb * 16 + l16, 512, kk, lh);
                v16bf bfr = load_frag_b(Ks, scb * 16 + l16, 512, kk, lh);
                s = wmma_bf16(afr, bfr, s);
            }
#pragma unroll
            for (int e = 0; e < 8; ++e) {
                int r = srb * 16 + e + 8 * lh;
                Sf[r * 32 + scb * 16 + l16] = s[e];
            }
        }
        __syncthreads();

        // ---- online softmax update (one lane per query row) ----
        if (tid < 64) {
            const int r = tid;
            float mOld = rowM[r];
            float mj = -INFINITY;
            for (int j = 0; j < 32; ++j) mj = fmaxf(mj, Sf[r * 32 + j]);
            float mNew = fmaxf(mOld, mj);
            float scale = __expf(mOld - mNew);
            float sum = 0.0f;
            unsigned* prow = (unsigned*)&Ps[r * 32];
            for (int j = 0; j < 32; j += 2) {
                float e0 = __expf(Sf[r * 32 + j]     - mNew);
                float e1 = __expf(Sf[r * 32 + j + 1] - mNew);
                sum += e0 + e1;
                prow[j >> 1] = pk_bf16(e0, e1);
            }
            rowM[r] = mNew;
            rowL[r] = rowL[r] * scale + sum;
            rowScale[r] = scale;
        }
        __syncthreads();

        // ---- rescale accumulators, then P*V and P*V^2 (K = 32, one WMMA each) ----
        float sc[8];
#pragma unroll
        for (int e = 0; e < 8; ++e) sc[e] = rowScale[rb * 16 + e + 8 * lh];
#pragma unroll
        for (int t = 0; t < 8; ++t) {
#pragma unroll
            for (int e = 0; e < 8; ++e) { accM[t][e] *= sc[e]; accS[t][e] *= sc[e]; }
        }
        v16bf pfr = load_frag_a(Ps, rb * 16 + l16, 32, 0, lh);
#pragma unroll
        for (int t = 0; t < 8; ++t) {
            int c = cbase + t * 16 + l16;
            v16bf bv  = load_frag_b(Vt,  c, 32, 0, lh);
            v16bf bv2 = load_frag_b(Vt2, c, 32, 0, lh);
            accM[t] = wmma_bf16(pfr, bv,  accM[t]);
            accS[t] = wmma_bf16(pfr, bv2, accS[t]);
        }
    }
    __syncthreads();

    // ---- epilogue: divide by softmax denom, std = sqrt(relu(E[x^2]-E[x]^2)) ----
    float inv[8];
#pragma unroll
    for (int e = 0; e < 8; ++e) inv[e] = 1.0f / rowL[rb * 16 + e + 8 * lh];
#pragma unroll
    for (int t = 0; t < 8; ++t) {
        int c = cbase + t * 16 + l16;
#pragma unroll
        for (int e = 0; e < 8; ++e) {
            int n = n0 + rb * 16 + e + 8 * lh;
            float mu = accM[t][e] * inv[e];
            float s2 = accS[t][e] * inv[e];
            float sd = sqrtf(fmaxf(s2 - mu * mu, 0.0f));
            size_t idx = ((size_t)b * N + n) * C + c;
            outMean[idx] = mu;
            outStd[idx]  = sd;
        }
    }
}

// ============================================================================
// Kernel 3: per-(b,c) mean / 1/sqrt(var+eps) of content (unbiased var, ddof=1)
// ============================================================================
__global__ __launch_bounds__(256) void mvn_stats_kernel(const float* __restrict__ content,
                                                        float* __restrict__ stats) {
    const int c = blockIdx.x, b = blockIdx.y;
    const float* p = content + ((size_t)b * C + c) * N;
    float s = 0.0f, ss = 0.0f;
    for (int i = threadIdx.x; i < N; i += 256) { float x = p[i]; s += x; ss += x * x; }
    __shared__ float rs[256], rss[256];
    rs[threadIdx.x] = s; rss[threadIdx.x] = ss;
    __syncthreads();
    for (int off = 128; off > 0; off >>= 1) {
        if (threadIdx.x < off) { rs[threadIdx.x] += rs[threadIdx.x + off];
                                 rss[threadIdx.x] += rss[threadIdx.x + off]; }
        __syncthreads();
    }
    if (threadIdx.x == 0) {
        float mean = rs[0] / (float)N;
        float var  = (rss[0] - (float)N * mean * mean) / (float)(N - 1);
        stats[((size_t)b * C + c) * 2 + 0] = mean;
        stats[((size_t)b * C + c) * 2 + 1] = 1.0f / sqrtf(var + 1e-5f);
    }
}

// ============================================================================
// Kernel 4: out[b,c,n] = attnStd[b,n,c] * (content - mu_c)*inv_c + attnMean[b,n,c]
// 32x32 LDS transpose tile so both global streams stay coalesced.
// ============================================================================
__global__ __launch_bounds__(256) void finalize_kernel(const float* __restrict__ content,
                                                       const float* __restrict__ attnMean,
                                                       const float* __restrict__ attnStd,
                                                       const float* __restrict__ stats,
                                                       float* __restrict__ out) {
    const int n0 = blockIdx.x * 32, c0 = blockIdx.y * 32, b = blockIdx.z;
    __shared__ float mT[32 * 32], sT[32 * 32];
    for (int i = threadIdx.x; i < 1024; i += 256) {
        int nl = i >> 5, cl = i & 31;
        size_t idx = ((size_t)b * N + n0 + nl) * C + c0 + cl;
        mT[i] = attnMean[idx];
        sT[i] = attnStd[idx];
    }
    __syncthreads();
    for (int i = threadIdx.x; i < 1024; i += 256) {
        int cl = i >> 5, nl = i & 31;
        int c = c0 + cl;
        float mu = stats[((size_t)b * C + c) * 2 + 0];
        float iv = stats[((size_t)b * C + c) * 2 + 1];
        size_t g = ((size_t)b * C + c) * N + n0 + nl;
        float x = content[g];
        out[g] = sT[nl * 32 + cl] * (x - mu) * iv + mT[nl * 32 + cl];
    }
}

// ============================================================================
extern "C" void kernel_launch(void* const* d_in, const int* in_sizes, int n_in,
                              void* d_out, int out_size, void* d_ws, size_t ws_size,
                              hipStream_t stream) {
    const float* content = (const float*)d_in[0];
    const float* style   = (const float*)d_in[1];
    const float* ckey    = (const float*)d_in[2];
    const float* skey    = (const float*)d_in[3];
    const float* Wf = (const float*)d_in[4];  const float* bf_ = (const float*)d_in[5];
    const float* Wg = (const float*)d_in[6];  const float* bg  = (const float*)d_in[7];
    const float* Wh = (const float*)d_in[8];  const float* bh  = (const float*)d_in[9];

    // workspace layout (bytes):
    //   F bf16 : 16 MB   G bf16 : 16 MB   V bf16 : 16 MB
    //   attnMean f32 : 32 MB   attnStd f32 : 32 MB   stats f32 : 16 KB
    char* wsp = (char*)d_ws;
    const size_t szb = (size_t)Bn * N * C * 2;
    const size_t szf = (size_t)Bn * N * C * 4;
    unsigned short* Fbf = (unsigned short*)(wsp);
    unsigned short* Gbf = (unsigned short*)(wsp + szb);
    unsigned short* Vbf = (unsigned short*)(wsp + 2 * szb);
    float* aMean = (float*)(wsp + 3 * szb);
    float* aStd  = (float*)(wsp + 3 * szb + szf);
    float* stats = (float*)(wsp + 3 * szb + 2 * szf);

    dim3 pg(N / 64, C / 64, Bn);
    proj_kernel<<<pg, 128, 0, stream>>>(ckey, Wf, bf_, Fbf);   // F = Wf*content_key + bf
    proj_kernel<<<pg, 128, 0, stream>>>(skey, Wg, bg, Gbf);    // G = Wg*style_key + bg
    proj_kernel<<<pg, 128, 0, stream>>>(style, Wh, bh, Vbf);   // V = Wh*style + bh

    attn_kernel<<<dim3(N / 64, Bn), 512, 0, stream>>>(Fbf, Gbf, Vbf, aMean, aStd);

    mvn_stats_kernel<<<dim3(C, Bn), 256, 0, stream>>>(content, stats);

    finalize_kernel<<<dim3(N / 32, C / 32, Bn), 256, 0, stream>>>(content, aMean, aStd,
                                                                  stats, (float*)d_out);
}
